// SwitchTransformersSparseMLP_29858612642048
// MI455X (gfx1250) — matrix operand order repaired and checked by reading, not compiled, not run on hardware
//
#include <hip/hip_runtime.h>
#include <hip/hip_bf16.h>

// Problem sizes (match reference)
static constexpr int B  = 8;
static constexpr int S  = 2048;
static constexpr int D  = 768;
static constexpr int FF = 3072;
static constexpr int E  = 8;
static constexpr int C  = 384;

typedef __attribute__((ext_vector_type(16))) __bf16 v16bf;
typedef __attribute__((ext_vector_type(8)))  __bf16 v8bf;
typedef __attribute__((ext_vector_type(8)))  float  v8f;

// ---------------------------------------------------------------------------
// WMMA operand fragments (bf16, 16x16x32).
//
// A operand (16x32, MxK): lanes 0-15 hold row M=lane with
// K = k..k+15 contiguous; lanes 16-31 hold K = k+16..k+31 (same order as B).
// => two 16-byte chunks per lane at (row base + sel16) + k and + k + 8,
// where sel16 = (lane>>4)*16.
//
// B operand (32x16, KxN): lane n<16 = column n with K = k..k+15 contiguous;
// lanes 16-31 = column n-16 with K = k+16..k+31.  => one 32-byte run per lane
// at (col base + sel16) + k, sel16 = (lane>>4)*16, col base walking a
// K-contiguous array.
//
// All tile offsets (i*16*K, t*16*K) are compile-time constants and fold into
// the instruction's 24-bit immediate offset: only TWO base pointers live in
// registers per GEMM (this is what kills the round-2 spills).
// ---------------------------------------------------------------------------
struct Frag {
    v16bf a[2];
    v16bf b[4];
};

template <int K>
static __device__ __forceinline__ Frag load_frag(const __bf16* __restrict__ aBase,
                                                 const __bf16* __restrict__ bBase,
                                                 int k) {
    Frag f;
    #pragma unroll
    for (int i = 0; i < 2; ++i) {
        const v8bf lo = *(const v8bf*)(aBase + i * 16 * K + k);
        const v8bf hi = *(const v8bf*)(aBase + i * 16 * K + k + 8);
        f.a[i] = __builtin_shufflevector(lo, hi, 0, 1, 2, 3, 4, 5, 6, 7,
                                                 8, 9, 10, 11, 12, 13, 14, 15);
    }
    #pragma unroll
    for (int t = 0; t < 4; ++t)
        f.b[t] = *(const v16bf*)(bBase + t * 16 * K + k);
    return f;
}

static __device__ __forceinline__ void mma_frag(v8f acc[2][4], const Frag& f) {
    #pragma unroll
    for (int i = 0; i < 2; ++i)
        #pragma unroll
        for (int t = 0; t < 4; ++t)
            acc[i][t] = __builtin_amdgcn_wmma_f32_16x16x32_bf16(
                false, f.a[i], false, f.b[t], (short)0, acc[i][t], false, false);
}

// Double-buffered K loop (K % 64 == 0), 32Mx64N wave tile, 16 WMMA / 64-K step.
template <int K>
static __device__ __forceinline__ void gemm_core(const __bf16* __restrict__ aBase,
                                                 const __bf16* __restrict__ bBase,
                                                 v8f acc[2][4]) {
    Frag f0 = load_frag<K>(aBase, bBase, 0);
    for (int k = 0; k < K; k += 64) {
        // WGP-scope prefetch ~4 iterations ahead; every enabled lane prefetches
        // its own row/column line, immediate offsets only.
        #pragma unroll
        for (int i = 0; i < 2; ++i)
            __builtin_prefetch(aBase + i * 16 * K + k + 256, 0, 3);
        #pragma unroll
        for (int t = 0; t < 4; ++t)
            __builtin_prefetch(bBase + t * 16 * K + k + 256, 0, 3);
        Frag f1 = load_frag<K>(aBase, bBase, k + 32);
        mma_frag(acc, f0);
        if (k + 64 < K) f0 = load_frag<K>(aBase, bBase, k + 64);
        mma_frag(acc, f1);
    }
}

// ---------------------------------------------------------------------------
// 1) Router: one wave (32 lanes) per token.
// ---------------------------------------------------------------------------
__global__ void router_kernel(const float* __restrict__ hs,
                              const float* __restrict__ Wr,
                              float* __restrict__ logits_out,
                              float* __restrict__ probmax,
                              int*   __restrict__ eidx) {
    const int tok  = (int)((blockIdx.x * blockDim.x + threadIdx.x) >> 5);
    const int lane = (int)(threadIdx.x & 31);
    if (tok >= B * S) return;

    const float* xr = hs + (size_t)tok * D;
    float acc[E] = {};
    for (int d = lane; d < D; d += 32) {
        const float xv = xr[d];
        const float* wr = Wr + (size_t)d * E;
        #pragma unroll
        for (int e = 0; e < E; ++e) acc[e] += xv * wr[e];
    }
    #pragma unroll
    for (int e = 0; e < E; ++e) {
        #pragma unroll
        for (int off = 16; off > 0; off >>= 1)
            acc[e] += __shfl_xor(acc[e], off, 32);
    }
    if (lane == 0) {
        float m = acc[0];
        int am = 0;
        #pragma unroll
        for (int e = 1; e < E; ++e)
            if (acc[e] > m) { m = acc[e]; am = e; }   // first-max, like jnp.argmax
        float sum = 0.0f;
        #pragma unroll
        for (int e = 0; e < E; ++e) sum += __expf(acc[e] - m);
        #pragma unroll
        for (int e = 0; e < E; ++e) logits_out[(size_t)tok * E + e] = acc[e];
        probmax[tok] = 1.0f / sum;                     // max softmax prob
        eidx[tok]    = am;
    }
}

// ---------------------------------------------------------------------------
// 2) Capacity scan: one thread per (b,e); order-sensitive cumsum over S.
// ---------------------------------------------------------------------------
__global__ void scan_kernel(const int* __restrict__ eidx,
                            int* __restrict__ slot_token,
                            int* __restrict__ eidx_out) {
    const int t = (int)threadIdx.x;
    if (t >= B * E) return;
    const int b = t / E, e = t % E;
    int* st = slot_token + (size_t)t * C;
    for (int c = 0; c < C; ++c) st[c] = -1;
    int cnt = 0;
    for (int s = 0; s < S; ++s) {
        if (eidx[b * S + s] == e) {
            const bool keep = (cnt < C);
            if (keep) st[cnt] = s;
            eidx_out[b * S + s] = keep ? e : 0;
            ++cnt;
        }
    }
}

// ---------------------------------------------------------------------------
// 3) Prefill out = probmax * x for ALL tokens (dropped tokens keep it).
// ---------------------------------------------------------------------------
__global__ void prefill_kernel(const float* __restrict__ hs,
                               const float* __restrict__ probmax,
                               float* __restrict__ out) {
    const size_t idx = (size_t)blockIdx.x * blockDim.x + threadIdx.x;
    if (idx >= (size_t)B * S * D) return;
    out[idx] = probmax[idx / D] * hs[idx];
}

// ---------------------------------------------------------------------------
// 4) Dispatch gather: build zero-padded xe_bf16 [B*E, C, D].
// ---------------------------------------------------------------------------
__global__ void dispatch_kernel(const float* __restrict__ hs,
                                const int* __restrict__ slot_token,
                                __bf16* __restrict__ xe) {
    const int slot = (int)blockIdx.x;            // 0 .. B*E*C-1
    const int tok  = slot_token[slot];
    const int b    = slot / (E * C);
    __bf16* dst = xe + (size_t)slot * D;
    if (tok >= 0) {
        const float* src = hs + ((size_t)b * S + tok) * D;
        for (int i = (int)threadIdx.x; i < D; i += (int)blockDim.x)
            dst[i] = (__bf16)src[i];
    } else {
        for (int i = (int)threadIdx.x; i < D; i += (int)blockDim.x)
            dst[i] = (__bf16)0.0f;
    }
}

// ---------------------------------------------------------------------------
// 5) Weight convert + transpose to bf16, LDS-tiled (coalesced both ways).
//    in: [R][Cc] f32 per expert -> out: [Cc][R] bf16 per expert.
// ---------------------------------------------------------------------------
template <int R, int Cc>
__global__ void transpose_convert_kernel(const float* __restrict__ in,
                                         __bf16* __restrict__ outp) {
    __shared__ float tile[32][33];
    constexpr int tilesPerRow    = Cc / 32;
    constexpr int tilesPerExpert = (R / 32) * tilesPerRow;
    const int gid = (int)blockIdx.x;
    const int e   = gid / tilesPerExpert;
    const int t   = gid % tilesPerExpert;
    const int tr  = t / tilesPerRow;
    const int tc  = t % tilesPerRow;
    const float* src = in   + (size_t)e * R * Cc;
    __bf16*      dst = outp + (size_t)e * R * Cc;
    const int x = (int)(threadIdx.x & 31);
    const int y = (int)(threadIdx.x >> 5);
    #pragma unroll
    for (int yy = y; yy < 32; yy += 8)
        tile[yy][x] = src[(size_t)(tr * 32 + yy) * Cc + tc * 32 + x];
    __syncthreads();
    #pragma unroll
    for (int yy = y; yy < 32; yy += 8)
        dst[(size_t)(tc * 32 + yy) * R + tr * 32 + x] = (__bf16)tile[x][yy];
}

// ---------------------------------------------------------------------------
// 6) GEMM1: h = relu(xe @ Wi).  M=C(384), N=FF(3072), K=D(768).
//    Block = 2 M-waves x 4 N-waves => 64M x 256N.
// ---------------------------------------------------------------------------
__global__ void gemm1_kernel(const __bf16* __restrict__ xe,
                             const __bf16* __restrict__ WiT,
                             __bf16* __restrict__ h) {
    const int be   = (int)blockIdx.z;
    const int e    = be % E;
    const int lane = (int)(threadIdx.x & 31);
    const int wid  = (int)(threadIdx.x >> 5);
    const int M0   = (int)blockIdx.y * 64 + (wid & 1) * 32;
    const int N0   = (int)blockIdx.x * 256 + (wid >> 1) * 64;

    const __bf16* aBase = xe + (size_t)be * C * D
                        + (size_t)(M0 + (lane & 15)) * D + (lane >> 4) * 16;
    const __bf16* bBase = WiT + (size_t)e * FF * D
                        + (size_t)(N0 + (lane & 15)) * D + (lane >> 4) * 16;

    v8f acc[2][4] = {};
    gemm_core<D>(aBase, bBase, acc);

    // Epilogue: ReLU + bf16 store. C/D layout: VGPR r holds
    // (M = r + 8*(lane>>4), N = lane&15) within each 16x16 tile.
    __bf16* hb = h + (size_t)be * C * FF;
    #pragma unroll
    for (int i = 0; i < 2; ++i) {
        const int rbase = M0 + i * 16 + 8 * (lane >> 4);
        #pragma unroll
        for (int t = 0; t < 4; ++t) {
            const int col = N0 + t * 16 + (lane & 15);
            #pragma unroll
            for (int r = 0; r < 8; ++r) {
                float v = acc[i][t][r];
                v = v > 0.0f ? v : 0.0f;
                hb[(size_t)(rbase + r) * FF + col] = (__bf16)v;
            }
        }
    }
}

// ---------------------------------------------------------------------------
// 7) GEMM2, TRANSPOSED: ye^T = Wo^T @ h^T.
//    M = D (768, rows of WoT), N = C (384, capacity slots), K = FF (3072).
//    A = WoT[e] ([D][FF] K-contiguous rows).  B = h[be] natural [C][FF]
//    layout (B-operand wants K-contiguous per column == a row of h).
//    Block = 4 M-waves x 2 N-waves => 128M x 128N.
//    Epilogue: lane owns (col=c -> one token) x 8 consecutive d -> float4 x2.
// ---------------------------------------------------------------------------
__global__ void gemm2_kernel(const __bf16* __restrict__ h,
                             const __bf16* __restrict__ WoT,
                             const int*   __restrict__ slot_token,
                             const float* __restrict__ probmax,
                             float* __restrict__ out) {
    const int be   = (int)blockIdx.z;
    const int b    = be / E;
    const int e    = be % E;
    const int lane = (int)(threadIdx.x & 31);
    const int wid  = (int)(threadIdx.x >> 5);
    const int M0   = (int)blockIdx.y * 128 + (wid & 3) * 32;   // d
    const int N0   = (int)blockIdx.x * 128 + (wid >> 2) * 64;  // capacity slot

    const __bf16* aBase = WoT + (size_t)e * D * FF
                        + (size_t)(M0 + (lane & 15)) * FF + (lane >> 4) * 16;
    const __bf16* bBase = h + (size_t)be * C * FF
                        + (size_t)(N0 + (lane & 15)) * FF + (lane >> 4) * 16;

    v8f acc[2][4] = {};
    gemm_core<FF>(aBase, bBase, acc);

    #pragma unroll
    for (int t = 0; t < 4; ++t) {
        const int c   = N0 + t * 16 + (lane & 15);     // capacity slot (column)
        const int tok = slot_token[be * C + c];
        if (tok < 0) continue;                         // unfilled slot
        const float scale = probmax[b * S + tok];
        float* orow = out + ((size_t)b * S + tok) * D;
        #pragma unroll
        for (int i = 0; i < 2; ++i) {
            const int dbase = M0 + i * 16 + 8 * (lane >> 4);
            float4 lo, hi;
            lo.x = scale * acc[i][t][0]; lo.y = scale * acc[i][t][1];
            lo.z = scale * acc[i][t][2]; lo.w = scale * acc[i][t][3];
            hi.x = scale * acc[i][t][4]; hi.y = scale * acc[i][t][5];
            hi.z = scale * acc[i][t][6]; hi.w = scale * acc[i][t][7];
            *(float4*)(orow + dbase)     = lo;
            *(float4*)(orow + dbase + 4) = hi;
        }
    }
}

// ---------------------------------------------------------------------------
// Launch
// ---------------------------------------------------------------------------
extern "C" void kernel_launch(void* const* d_in, const int* in_sizes, int n_in,
                              void* d_out, int out_size, void* d_ws, size_t ws_size,
                              hipStream_t stream) {
    (void)in_sizes; (void)n_in; (void)out_size; (void)ws_size;

    const float* hs = (const float*)d_in[0];   // [B,S,D]
    const float* Wr = (const float*)d_in[1];   // [D,E]
    const float* Wi = (const float*)d_in[2];   // [E,D,FF]
    const float* Wo = (const float*)d_in[3];   // [E,FF,D]

    // Outputs concatenated: out [B,S,D] f32, router_logits [B,S,E] f32,
    // expert_index [B,S] int32 (trailing region).
    float* out      = (float*)d_out;
    float* logits   = out + (size_t)B * S * D;
    int*   eidx_out = (int*)(out + (size_t)B * S * D + (size_t)B * S * E);

    // Workspace carve-up (all region sizes are multiples of 256 bytes).
    char* ws = (char*)d_ws;
    float*  probmax    = (float*)ws;  ws += (size_t)B * S * sizeof(float);        //  64 KB
    int*    eidx       = (int*)ws;    ws += (size_t)B * S * sizeof(int);          //  64 KB
    int*    slot_token = (int*)ws;    ws += (size_t)B * E * C * sizeof(int);      //  96 KB
    __bf16* xe         = (__bf16*)ws; ws += (size_t)B * E * C * D * 2;            // 37.7 MB
    __bf16* WiT        = (__bf16*)ws; ws += (size_t)E * FF * D * 2;               // 37.7 MB
    __bf16* WoT        = (__bf16*)ws; ws += (size_t)E * D * FF * 2;               // 37.7 MB
    __bf16* hbuf       = (__bf16*)ws;                                             // 151 MB

    router_kernel<<<(B * S) / 8, 256, 0, stream>>>(hs, Wr, logits, probmax, eidx);
    scan_kernel<<<1, 64, 0, stream>>>(eidx, slot_token, eidx_out);
    prefill_kernel<<<(int)(((size_t)B * S * D) / 256), 256, 0, stream>>>(hs, probmax, out);
    dispatch_kernel<<<B * E * C, 256, 0, stream>>>(hs, slot_token, xe);
    transpose_convert_kernel<D, FF><<<E * (D / 32) * (FF / 32), 256, 0, stream>>>(Wi, WiT);
    transpose_convert_kernel<FF, D><<<E * (FF / 32) * (D / 32), 256, 0, stream>>>(Wo, WoT);
    dim3 g1(FF / 256, C / 64, B * E);
    gemm1_kernel<<<g1, 256, 0, stream>>>(xe, WiT, hbuf);
    dim3 g2(C / 128, D / 128, B * E);
    gemm2_kernel<<<g2, 256, 0, stream>>>(hbuf, WoT, slot_token, probmax, out);
}
